// TransformerBlock_30992484008185
// MI455X (gfx1250) — matrix operand order repaired and checked by reading, not compiled, not run on hardware
//
#include <hip/hip_runtime.h>
#include <hip/hip_bf16.h>

// ---------------------------------------------------------------------------
// Transformer block for MI455X (gfx1250): fp32 I/O, bf16 WMMA compute path.
//   v_wmma_f32_16x16x32_bf16 for all GEMMs + flash attention (no score tensor).
//   Async global->LDS staging (ASYNCcnt) when the toolchain exposes it,
//   global_prefetch_b8 for next-tile prefetch.
// ---------------------------------------------------------------------------

typedef __bf16 bf16;
typedef __bf16 v16bf __attribute__((ext_vector_type(16)));
typedef float  v8f   __attribute__((ext_vector_type(8)));
typedef int    v4i   __attribute__((vector_size(16)));   // matches builtin proto

#define L_SEQ 4096
#define EMB   1024
#define HID   4096
#define HEADS 16
#define HDIM  64

#define AS1 __attribute__((address_space(1)))
#define AS3 __attribute__((address_space(3)))

#if defined(__has_builtin)
#if __has_builtin(__builtin_amdgcn_global_load_async_to_lds_b128) && \
    __has_builtin(__builtin_amdgcn_s_wait_asynccnt)
#define HAVE_ASYNC_LDS 1
#endif
#endif

union BF16Frag {           // 16 bf16 = one WMMA A/B operand (8 VGPRs)
    v16bf v;
    uint4 q[2];
};

__device__ __forceinline__ v8f wmma_bf16(v16bf a, v16bf b, v8f c) {
    // (neg_a, A, neg_b, B, c_mod, C, reuse_a, reuse_b)
    return __builtin_amdgcn_wmma_f32_16x16x32_bf16(false, a, false, b, (short)0, c,
                                                   false, false);
}

// 16B global -> LDS copy; async DMA path on gfx1250 toolchains that expose it.
__device__ __forceinline__ void g2l_b128(bf16* lds, const bf16* g) {
#if defined(HAVE_ASYNC_LDS)
    __builtin_amdgcn_global_load_async_to_lds_b128((AS1 v4i*)g, (AS3 v4i*)lds,
                                                   0, 0);
#else
    *(uint4*)lds = *(const uint4*)g;
#endif
}
__device__ __forceinline__ void wait_async_copies() {
#if defined(HAVE_ASYNC_LDS)
    __builtin_amdgcn_s_wait_asynccnt(0);
#endif
}

// ---------------------------------------------------------------------------
// fp32 -> bf16 downconvert (n must be a multiple of 1024)
// ---------------------------------------------------------------------------
__global__ __launch_bounds__(256) void cvt_kernel(const float* __restrict__ s,
                                                  bf16* __restrict__ d, int n) {
    int i = (blockIdx.x * 256 + threadIdx.x) * 4;
    if (i + 3 < n) {
        float4 v = *(const float4*)(s + i);
        bf16 t[4];
        t[0] = (bf16)v.x; t[1] = (bf16)v.y; t[2] = (bf16)v.z; t[3] = (bf16)v.w;
        *(uint2*)(d + i) = *(const uint2*)t;   // 8B packed store
    }
}

// ---------------------------------------------------------------------------
// C[M,N] = act(A[M,K] @ B[K,N] + bias)   (A,B bf16 row-major; M%128==N%64==K%32==0)
// Block: 128 threads (4 waves). Block tile 128x64, BK=32.
// Wave w owns rows [w*32, w*32+32): 2 M-subtiles x 4 N-subtiles = 8 WMMA/K-step.
// ---------------------------------------------------------------------------
__global__ __launch_bounds__(128) void gemm_bf16_kernel(
    const bf16* __restrict__ A, const bf16* __restrict__ B,
    const float* __restrict__ bias,
    float* __restrict__ Cf, bf16* __restrict__ Cb,
    int M, int N, int K, int gelu) {
    __shared__ bf16 sA[128 * 32];   // [m][k] row-major
    __shared__ bf16 sBt[64 * 32];   // transposed: [n][k]

    const int tid  = threadIdx.x;
    const int lane = tid & 31;
    const int wave = tid >> 5;
    const int half = lane >> 4;
    const int l16  = lane & 15;

    const int mBlock = blockIdx.y * 128;
    const int nBlock = blockIdx.x * 64;

    v8f acc[2][4];
#pragma unroll
    for (int mt = 0; mt < 2; ++mt)
#pragma unroll
        for (int nt = 0; nt < 4; ++nt) acc[mt][nt] = 0.f;

    for (int k0 = 0; k0 < K; k0 += 32) {
        __syncthreads();
        // --- A tile: 128x32 bf16 via async 16B global->LDS DMA ---
#pragma unroll
        for (int i = 0; i < 4; ++i) {
            int c   = tid + i * 128;          // 0..511 chunks of 16B
            int row = c >> 2;
            int off = (c & 3) * 8;
            g2l_b128(sA + row * 32 + off,
                     A + (size_t)(mBlock + row) * K + k0 + off);
        }
        // --- B tile: 32x64 bf16, stored transposed sBt[n][k], packed b32 ---
        {
            int kb = (tid >> 5) * 2;          // 0,2,4,6
            int n2 = (tid & 31) * 2;          // even column
#pragma unroll
            for (int kk = 0; kk < 32; kk += 8) {
                int k = kk + kb;
                unsigned int w0 =
                    *(const unsigned int*)(B + (size_t)(k0 + k) * N + nBlock + n2);
                unsigned int w1 =
                    *(const unsigned int*)(B + (size_t)(k0 + k + 1) * N + nBlock + n2);
                unsigned int lo = (w0 & 0xFFFFu) | (w1 << 16);       // col n2
                unsigned int hi = (w0 >> 16) | (w1 & 0xFFFF0000u);   // col n2+1
                *(unsigned int*)(sBt + (n2 + 0) * 32 + k) = lo;
                *(unsigned int*)(sBt + (n2 + 1) * 32 + k) = hi;
            }
        }
        // prefetch next K-tile while this one is consumed
        if (k0 + 32 < K) {
            __builtin_prefetch(A + (size_t)(mBlock + tid) * K + k0 + 32, 0, 1);
            __builtin_prefetch(B + (size_t)(k0 + 32 + (tid >> 2)) * N + nBlock, 0, 1);
        }
        wait_async_copies();
        __syncthreads();

        // A fragments: lane holds row m; K = half*16 + e (contiguous, same order as B)
        BF16Frag af[2];
#pragma unroll
        for (int mt = 0; mt < 2; ++mt) {
            const bf16* p = sA + (wave * 32 + mt * 16 + l16) * 32;
            af[mt].q[0] = *(const uint4*)(p + half * 16);
            af[mt].q[1] = *(const uint4*)(p + half * 16 + 8);
        }
#pragma unroll
        for (int nt = 0; nt < 4; ++nt) {
            BF16Frag bfrag;                    // lane = column n, K = half*16 + e
            const bf16* p = sBt + (nt * 16 + l16) * 32 + half * 16;
            bfrag.q[0] = *(const uint4*)p;
            bfrag.q[1] = *(const uint4*)(p + 8);
#pragma unroll
            for (int mt = 0; mt < 2; ++mt)
                acc[mt][nt] = wmma_bf16(af[mt].v, bfrag.v, acc[mt][nt]);
        }
    }

    // Epilogue: C layout row = r + 8*half, col = l16
#pragma unroll
    for (int mt = 0; mt < 2; ++mt) {
        const int m0 = mBlock + wave * 32 + mt * 16;
#pragma unroll
        for (int nt = 0; nt < 4; ++nt) {
            int n = nBlock + nt * 16 + l16;
            float bs = bias[n];
#pragma unroll
            for (int r = 0; r < 8; ++r) {
                int m = m0 + r + 8 * half;
                float v = acc[mt][nt][r] + bs;
                if (gelu) v = 0.5f * v * (1.0f + erff(v * 0.70710678118f));
                if (Cf) Cf[(size_t)m * N + n] = v;
                if (Cb) Cb[(size_t)m * N + n] = (bf16)v;
            }
        }
    }
}

// ---------------------------------------------------------------------------
// Flash attention. Grid: (L/64 query blocks, HEADS). Block: 128 threads.
// Wave owns 16 query rows; streams 32-key blocks with online softmax.
// Q/K/V are bf16 [L][EMB], head h occupies columns h*64..h*64+63.
// ---------------------------------------------------------------------------
__global__ __launch_bounds__(128) void attn_kernel(
    const bf16* __restrict__ Qb, const bf16* __restrict__ Kb,
    const bf16* __restrict__ Vb, const int* __restrict__ mask,
    bf16* __restrict__ ctx) {
    const int h    = blockIdx.y;
    const int tid  = threadIdx.x;
    const int lane = tid & 31;
    const int wave = tid >> 5;
    const int half = lane >> 4;
    const int l16  = lane & 15;
    const int qBase = blockIdx.x * 64 + wave * 16;
    const int hoff  = h * HDIM;

    __shared__ bf16 sVt[64 * 32];      // V^T tile: [d][key]  (shared by all waves)
    __shared__ bf16 sP[4][16 * 32];    // per-wave P scratch (C-layout -> A-layout)

    // Q fragments (K-dim 64 = 2 chunks of 32), resident for the whole kernel.
    BF16Frag qf[2];
#pragma unroll
    for (int kc = 0; kc < 2; ++kc) {
        const bf16* p = Qb + (size_t)(qBase + l16) * EMB + hoff + kc * 32;
        qf[kc].q[0] = *(const uint4*)(p + half * 16);
        qf[kc].q[1] = *(const uint4*)(p + half * 16 + 8);
    }

    v8f o[4];
#pragma unroll
    for (int i = 0; i < 4; ++i) o[i] = 0.f;
    float mrun[8], lrun[8];
#pragma unroll
    for (int r = 0; r < 8; ++r) { mrun[r] = -3.0e38f; lrun[r] = 0.f; }

    for (int j = 0; j < L_SEQ; j += 32) {
        __syncthreads();
        // V tile (32 keys x 64 d) -> sVt[d][key], packed b32 transpose
        {
            int kb = (tid >> 5) * 2;
            int d2 = (tid & 31) * 2;
#pragma unroll
            for (int kk = 0; kk < 32; kk += 8) {
                int k = kk + kb;
                unsigned int w0 =
                    *(const unsigned int*)(Vb + (size_t)(j + k) * EMB + hoff + d2);
                unsigned int w1 =
                    *(const unsigned int*)(Vb + (size_t)(j + k + 1) * EMB + hoff + d2);
                unsigned int lo = (w0 & 0xFFFFu) | (w1 << 16);
                unsigned int hi = (w0 >> 16) | (w1 & 0xFFFF0000u);
                *(unsigned int*)(sVt + (d2 + 0) * 32 + k) = lo;
                *(unsigned int*)(sVt + (d2 + 1) * 32 + k) = hi;
            }
        }
        if (j + 32 < L_SEQ) {
            __builtin_prefetch(Kb + (size_t)(j + 32 + (tid & 31)) * EMB + hoff, 0, 1);
            __builtin_prefetch(Vb + (size_t)(j + 32 + (tid & 31)) * EMB + hoff, 0, 1);
        }
        __syncthreads();

        // S = Q K^T  (K row-major == B^T layout: load B frags straight from global)
        v8f s[2];
#pragma unroll
        for (int sub = 0; sub < 2; ++sub) {
            v8f a = 0.f;
#pragma unroll
            for (int kc = 0; kc < 2; ++kc) {
                BF16Frag kf;
                const bf16* p =
                    Kb + (size_t)(j + sub * 16 + l16) * EMB + hoff + kc * 32 + half * 16;
                kf.q[0] = *(const uint4*)p;
                kf.q[1] = *(const uint4*)(p + 8);
                a = wmma_bf16(qf[kc].v, kf.v, a);
            }
            s[sub] = a;
        }
        // scale + mask
#pragma unroll
        for (int sub = 0; sub < 2; ++sub)
#pragma unroll
            for (int r = 0; r < 8; ++r) {
                int q   = qBase + r + 8 * half;
                int key = j + sub * 16 + l16;
                float v = s[sub][r] * 0.125f;   // 1/sqrt(64)
                if (mask[(size_t)q * L_SEQ + key] == 0) v = -1.0e9f;
                s[sub][r] = v;
            }
        // online softmax: rows live in 16-lane half-groups (xor over bits 0..3)
        float alpha[8], mnew[8];
#pragma unroll
        for (int r = 0; r < 8; ++r) {
            float mx = fmaxf(s[0][r], s[1][r]);
#pragma unroll
            for (int off = 1; off < 16; off <<= 1)
                mx = fmaxf(mx, __shfl_xor(mx, off, 32));
            mnew[r]  = fmaxf(mrun[r], mx);
            alpha[r] = __expf(mrun[r] - mnew[r]);
            mrun[r]  = mnew[r];
        }
#pragma unroll
        for (int r = 0; r < 8; ++r) {
            float p0 = __expf(s[0][r] - mnew[r]);
            float p1 = __expf(s[1][r] - mnew[r]);
            s[0][r] = p0; s[1][r] = p1;
            float ls = p0 + p1;
#pragma unroll
            for (int off = 1; off < 16; off <<= 1)
                ls += __shfl_xor(ls, off, 32);
            lrun[r] = lrun[r] * alpha[r] + ls;
        }
        // P: C-layout -> wave-private LDS -> A-layout fragment
        bf16* P = sP[wave];
#pragma unroll
        for (int sub = 0; sub < 2; ++sub)
#pragma unroll
            for (int r = 0; r < 8; ++r)
                P[(r + 8 * half) * 32 + sub * 16 + l16] = (bf16)s[sub][r];
#pragma unroll
        for (int nt = 0; nt < 4; ++nt)
#pragma unroll
            for (int r = 0; r < 8; ++r) o[nt][r] *= alpha[r];

        BF16Frag pf;
        {
            const bf16* p = P + l16 * 32;
            pf.q[0] = *(const uint4*)(p + half * 16);
            pf.q[1] = *(const uint4*)(p + half * 16 + 8);
        }
#pragma unroll
        for (int nt = 0; nt < 4; ++nt) {
            BF16Frag vf;
            const bf16* p = sVt + (nt * 16 + l16) * 32 + half * 16;
            vf.q[0] = *(const uint4*)p;
            vf.q[1] = *(const uint4*)(p + 8);
            o[nt] = wmma_bf16(pf.v, vf.v, o[nt]);
        }
    }

    // normalize and emit ctx (bf16, feeds Wo GEMM)
#pragma unroll
    for (int nt = 0; nt < 4; ++nt)
#pragma unroll
        for (int r = 0; r < 8; ++r) {
            int q = qBase + r + 8 * half;
            int d = nt * 16 + l16;
            ctx[(size_t)q * EMB + hoff + d] = (bf16)(o[nt][r] / lrun[r]);
        }
}

// ---------------------------------------------------------------------------
// out = LayerNorm(X + Y) * g + b ; dual f32/bf16 outputs. One row per block.
// ---------------------------------------------------------------------------
__global__ __launch_bounds__(256) void add_ln_kernel(
    const float* __restrict__ X, const float* __restrict__ Y,
    const float* __restrict__ g, const float* __restrict__ b,
    float* __restrict__ outF, bf16* __restrict__ outB) {
    __shared__ float red[8];
    const int row = blockIdx.x;
    const float* x = X + (size_t)row * EMB;
    const float* y = Y + (size_t)row * EMB;

    float v[4];
    float sum = 0.f;
#pragma unroll
    for (int i = 0; i < 4; ++i) {
        v[i] = x[threadIdx.x + i * 256] + y[threadIdx.x + i * 256];
        sum += v[i];
    }
#pragma unroll
    for (int off = 1; off < 32; off <<= 1) sum += __shfl_xor(sum, off, 32);
    if ((threadIdx.x & 31) == 0) red[threadIdx.x >> 5] = sum;
    __syncthreads();
    float tot = 0.f;
#pragma unroll
    for (int w = 0; w < 8; ++w) tot += red[w];
    float mu = tot * (1.0f / EMB);
    __syncthreads();

    float vs = 0.f;
#pragma unroll
    for (int i = 0; i < 4; ++i) { float d = v[i] - mu; vs += d * d; }
#pragma unroll
    for (int off = 1; off < 32; off <<= 1) vs += __shfl_xor(vs, off, 32);
    if ((threadIdx.x & 31) == 0) red[threadIdx.x >> 5] = vs;
    __syncthreads();
    float tv = 0.f;
#pragma unroll
    for (int w = 0; w < 8; ++w) tv += red[w];
    float rstd = rsqrtf(tv * (1.0f / EMB) + 1e-5f);

#pragma unroll
    for (int i = 0; i < 4; ++i) {
        int c = threadIdx.x + i * 256;
        float o = (v[i] - mu) * rstd * g[c] + b[c];
        if (outF) outF[(size_t)row * EMB + c] = o;
        if (outB) outB[(size_t)row * EMB + c] = (bf16)o;
    }
}

// ---------------------------------------------------------------------------
extern "C" void kernel_launch(void* const* d_in, const int* in_sizes, int n_in,
                              void* d_out, int out_size, void* d_ws, size_t ws_size,
                              hipStream_t stream) {
    const float* x   = (const float*)d_in[0];
    const int*   msk = (const int*)d_in[1];
    const float* Wq  = (const float*)d_in[2];
    const float* bq  = (const float*)d_in[3];
    const float* Wk  = (const float*)d_in[4];
    const float* bk  = (const float*)d_in[5];
    const float* Wv  = (const float*)d_in[6];
    const float* bv  = (const float*)d_in[7];
    const float* Wo  = (const float*)d_in[8];
    const float* bo  = (const float*)d_in[9];
    const float* g1  = (const float*)d_in[10];
    const float* b1  = (const float*)d_in[11];
    const float* W1  = (const float*)d_in[12];
    const float* bf1 = (const float*)d_in[13];
    const float* W2  = (const float*)d_in[14];
    const float* bf2 = (const float*)d_in[15];
    const float* g2  = (const float*)d_in[16];
    const float* b2  = (const float*)d_in[17];
    float* out = (float*)d_out;

    size_t off = 0;
    auto carve = [&](size_t bytes) -> void* {
        void* p = (char*)d_ws + off;
        off += (bytes + 255) & ~(size_t)255;
        return p;
    };
    const size_t LE2 = (size_t)L_SEQ * EMB * 2, LE4 = (size_t)L_SEQ * EMB * 4;
    bf16* xb   = (bf16*)carve(LE2);
    bf16* wqb  = (bf16*)carve((size_t)EMB * EMB * 2);
    bf16* wkb  = (bf16*)carve((size_t)EMB * EMB * 2);
    bf16* wvb  = (bf16*)carve((size_t)EMB * EMB * 2);
    bf16* wob  = (bf16*)carve((size_t)EMB * EMB * 2);
    bf16* w1b  = (bf16*)carve((size_t)EMB * HID * 2);
    bf16* w2b  = (bf16*)carve((size_t)HID * EMB * 2);
    bf16* Qb   = (bf16*)carve(LE2);
    bf16* Kb   = (bf16*)carve(LE2);
    bf16* Vb   = (bf16*)carve(LE2);
    bf16* ctxb = (bf16*)carve(LE2);
    float* aout = (float*)carve(LE4);
    float* y1f  = (float*)carve(LE4);
    bf16*  y1b  = (bf16*)carve(LE2);
    bf16*  hb   = (bf16*)carve((size_t)L_SEQ * HID * 2);
    float* ffn  = (float*)carve(LE4);

    auto cvt = [&](const float* s, bf16* d, int n) {
        cvt_kernel<<<n / 1024, 256, 0, stream>>>(s, d, n);
    };
    cvt(x,  xb,  L_SEQ * EMB);
    cvt(Wq, wqb, EMB * EMB);
    cvt(Wk, wkb, EMB * EMB);
    cvt(Wv, wvb, EMB * EMB);
    cvt(Wo, wob, EMB * EMB);
    cvt(W1, w1b, EMB * HID);
    cvt(W2, w2b, HID * EMB);

    dim3 blk(128);
    // Q/K/V projections: [4096,1024] @ [1024,1024]
    dim3 gProj(EMB / 64, L_SEQ / 128);
    gemm_bf16_kernel<<<gProj, blk, 0, stream>>>(xb, wqb, bq, nullptr, Qb,
                                                L_SEQ, EMB, EMB, 0);
    gemm_bf16_kernel<<<gProj, blk, 0, stream>>>(xb, wkb, bk, nullptr, Kb,
                                                L_SEQ, EMB, EMB, 0);
    gemm_bf16_kernel<<<gProj, blk, 0, stream>>>(xb, wvb, bv, nullptr, Vb,
                                                L_SEQ, EMB, EMB, 0);
    // attention
    attn_kernel<<<dim3(L_SEQ / 64, HEADS), blk, 0, stream>>>(Qb, Kb, Vb, msk, ctxb);
    // output projection -> f32
    gemm_bf16_kernel<<<gProj, blk, 0, stream>>>(ctxb, wob, bo, aout, nullptr,
                                                L_SEQ, EMB, EMB, 0);
    // x = LN(x + attn_out)
    add_ln_kernel<<<L_SEQ, 256, 0, stream>>>(x, aout, g1, b1, y1f, y1b);
    // h = gelu(x @ W1 + bf1) -> bf16
    gemm_bf16_kernel<<<dim3(HID / 64, L_SEQ / 128), blk, 0, stream>>>(
        y1b, w1b, bf1, nullptr, hb, L_SEQ, HID, EMB, 1);
    // ffn = h @ W2 + bf2 -> f32
    gemm_bf16_kernel<<<gProj, blk, 0, stream>>>(hb, w2b, bf2, ffn, nullptr,
                                                L_SEQ, EMB, HID, 0);
    // out = LN(y1 + ffn)
    add_ln_kernel<<<L_SEQ, 256, 0, stream>>>(y1f, ffn, g2, b2, out, nullptr);
}
